// Views2Points_4303557230959
// MI455X (gfx1250) — matrix operand order repaired and checked by reading, not compiled, not run on hardware
//
#include <hip/hip_runtime.h>

typedef __attribute__((ext_vector_type(16))) _Float16 v16h;
typedef __attribute__((ext_vector_type(8)))  _Float16 v8h;
typedef __attribute__((ext_vector_type(8)))  float    v8f;

// ---------------------------------------------------------------------------
// Conv3D (3x3x3, pad=1) as implicit GEMM via v_wmma_f32_16x16x32_f16.
// Activations channel-last: (B, D,H,W, C) f16.  Weights pre-packed into
// A-fragment order.  Each wave computes NT 16(M)x16(N) output tiles that
// share one A fragment per K-step.  The inner loop is phased: all NT B
// fragments are loaded+masked into distinct registers first, then NT WMMAs
// issue back-to-back — keeping the WMMA->VALU WAR hazard window free of the
// fragment-building VALU ops (no forced v_nops) and letting the scheduler
// overlap the next iteration's loads with the WMMA block.
// Halo handling is branchless: loads use clamped in-bounds addresses and the
// fragment is zero-selected per dword (v_cndmask), avoiding EXEC branching.
// VIEWS=1: input is packed 2D view features (B,64,64,96); the 3D broadcast
// (side/front/top) is applied on the fly in the B-fragment address.
// ---------------------------------------------------------------------------
template <int VIEWS, int NT>
__global__ __launch_bounds__(128)
void conv3d_wmma_k(const _Float16* __restrict__ in, const _Float16* __restrict__ wpk,
                   const float* __restrict__ bias, _Float16* __restrict__ out,
                   int Cin, int Cout, int D, int H, int W, int relu)
{
    const int lane = threadIdx.x & 31;
    const int wave = threadIdx.x >> 5;
    const int HW   = H * W;
    const int DHW  = D * HW;
    const int b    = blockIdx.z;
    const int mt   = blockIdx.y;
    const int nl   = lane & 15;    // N column within tile
    const int hv   = lane >> 4;    // half-wave selector
    const int CT   = Cin >> 5;     // 32-channel K tiles

    int nv[NT], dA[NT], hA[NT], wA[NT];
#pragma unroll
    for (int t = 0; t < NT; ++t) {
        int n = ((blockIdx.x * 4 + wave) * NT + t) * 16 + nl;
        nv[t] = n;
        dA[t] = n / HW;
        hA[t] = (n / W) % H;
        wA[t] = n % W;
    }

    v8f zf = {};
    v8f acc[NT];
#pragma unroll
    for (int t = 0; t < NT; ++t) acc[t] = zf;

    const long aBase = (long)mt * 27 * CT * 512;
    const long inB   = VIEWS ? ((long)b * 4096 * 96) : ((long)b * DHW * Cin);

    union Frag { v16h h; int i[8]; };

    for (int kpos = 0; kpos < 27; ++kpos) {
        const int kd = kpos / 9, kh = (kpos / 3) % 3, kw = kpos % 3;
        bool val[NT];
        long voff[NT];
        int  czd[NT], czh[NT], czw[NT];
#pragma unroll
        for (int t = 0; t < NT; ++t) {
            const int zd = dA[t] + kd - 1, zh = hA[t] + kh - 1, zw = wA[t] + kw - 1;
            val[t] = (zd >= 0) && (zd < D) && (zh >= 0) && (zh < H) &&
                     (zw >= 0) && (zw < W);
            const int cd = min(max(zd, 0), D - 1);
            const int ch = min(max(zh, 0), H - 1);
            const int cw = min(max(zw, 0), W - 1);
            if (VIEWS) { czd[t] = cd; czh[t] = ch; czw[t] = cw; }
            else       voff[t] = ((long)cd * HW + (long)ch * W + cw) * Cin;
        }
        for (int cit = 0; cit < CT; ++cit) {
            const long aoff = aBase + ((long)kpos * CT + cit) * 512 + lane * 16;
            v16h afrag = *(const v16h*)(wpk + aoff);

            // Phase 1: load + mask all NT B fragments into distinct registers.
            Frag bf[NT];
#pragma unroll
            for (int t = 0; t < NT; ++t) {
                long off;
                if (VIEWS) {
                    // ctile 0: side(h,w)  1: front(d,w)  2: top(d,h)
                    const int u  = (cit == 0) ? czh[t] : czd[t];
                    const int vv = (cit == 2) ? czh[t] : czw[t];
                    off = ((long)(u * 64 + vv)) * 96 + cit * 32 + 16 * hv;
                } else {
                    off = voff[t] + cit * 32 + 16 * hv;
                }
                Frag ld;
                ld.h = *(const v16h*)(in + inB + off);   // always in-bounds (clamped)
#pragma unroll
                for (int j = 0; j < 8; ++j) bf[t].i[j] = val[t] ? ld.i[j] : 0;
            }
            // Phase 2: NT back-to-back WMMAs sharing one A fragment.
#pragma unroll
            for (int t = 0; t < NT; ++t)
                acc[t] = __builtin_amdgcn_wmma_f32_16x16x32_f16(false, afrag, false, bf[t].h,
                                                                (short)0, acc[t], false, false);
        }
    }

    const int mbase = mt * 16 + hv * 8;
#pragma unroll
    for (int t = 0; t < NT; ++t) {
        const long obase = ((long)b * DHW + nv[t]) * Cout;
        if ((Cout & 15) == 0) {
            v8h ov;
#pragma unroll
            for (int j = 0; j < 8; ++j) {
                float v = acc[t][j] + bias[mbase + j];
                if (relu) v = fmaxf(v, 0.f);
                ov[j] = (_Float16)v;
            }
            *(v8h*)(out + obase + mbase) = ov;        // 16B contiguous per lane
        } else {                                      // Cout==2 (d5)
#pragma unroll
            for (int j = 0; j < 8; ++j) {
                const int m = mbase + j;
                if (m < Cout) {
                    float v = acc[t][j] + bias[m];
                    if (relu) v = fmaxf(v, 0.f);
                    out[obase + m] = (_Float16)v;
                }
            }
        }
    }
}

// Pack f32 OIDHW weights into f16 A-fragment order:
// [MT][27 kpos][CT][32 lanes][16 halves]; lane%16=M row, lane/16 = K half,
// K(e) = e + 16*half   (contiguous K order, matching the B fragment loads).
__global__ void pack_w_k(const float* __restrict__ w, _Float16* __restrict__ wpk,
                         int Cout, int Cin, long total)
{
    long i = (long)blockIdx.x * blockDim.x + threadIdx.x;
    if (i >= total) return;
    const int CT = Cin >> 5;
    int e    = (int)(i & 15); long t = i >> 4;
    int lane = (int)(t & 31); t >>= 5;
    int cit  = (int)(t % CT); t /= CT;
    int kpos = (int)(t % 27); t /= 27;
    int mt   = (int)t;
    int hvv  = lane >> 4, mrow = lane & 15;
    int K    = e + 16 * hvv;
    int m    = mt * 16 + mrow;
    int ci   = cit * 32 + K;
    float v  = (m < Cout) ? w[((long)m * Cin + ci) * 27 + kpos] : 0.f;
    wpk[i]   = (_Float16)v;
}

// ---------------- 2x2x2 max pool, channel-last f16 ----------------
__global__ void maxpool_k(const _Float16* __restrict__ in, _Float16* __restrict__ out,
                          int C, int OD, int OH, int OW, long total)
{
    long i = (long)blockIdx.x * blockDim.x + threadIdx.x;
    if (i >= total) return;
    int c  = (int)(i % C); long t = i / C;
    int ow = (int)(t % OW); t /= OW;
    int oh = (int)(t % OH); t /= OH;
    int od = (int)(t % OD); int b = (int)(t / OD);
    const int ID = OD * 2, IH = OH * 2, IW = OW * 2;
    float m = -3.4e38f;
#pragma unroll
    for (int dz = 0; dz < 2; ++dz)
#pragma unroll
        for (int dy = 0; dy < 2; ++dy)
#pragma unroll
            for (int dx = 0; dx < 2; ++dx) {
                long idx = ((((long)b * ID + (od * 2 + dz)) * IH + (oh * 2 + dy)) * IW
                            + (ow * 2 + dx)) * C + c;
                m = fmaxf(m, (float)in[idx]);
            }
    out[i] = (_Float16)m;
}

// ------- trilinear x2 upsample (half-pixel) + relu + optional skip add -------
__global__ void up2_k(const _Float16* __restrict__ in, const _Float16* __restrict__ skip,
                      _Float16* __restrict__ out, int C, int ID, int IH, int IW, long total)
{
    long i = (long)blockIdx.x * blockDim.x + threadIdx.x;
    if (i >= total) return;
    const int OD = 2 * ID, OH = 2 * IH, OW = 2 * IW;
    int c  = (int)(i % C); long t = i / C;
    int ow = (int)(t % OW); t /= OW;
    int oh = (int)(t % OH); t /= OH;
    int od = (int)(t % OD); int b = (int)(t / OD);

    float sd = (od + 0.5f) * 0.5f - 0.5f;
    float sh = (oh + 0.5f) * 0.5f - 0.5f;
    float sw = (ow + 0.5f) * 0.5f - 0.5f;
    int d0 = (int)floorf(sd); float fd = sd - d0;
    int h0 = (int)floorf(sh); float fh = sh - h0;
    int w0 = (int)floorf(sw); float fw = sw - w0;
    int d1 = min(max(d0 + 1, 0), ID - 1); d0 = min(max(d0, 0), ID - 1);
    int h1 = min(max(h0 + 1, 0), IH - 1); h0 = min(max(h0, 0), IH - 1);
    int w1 = min(max(w0 + 1, 0), IW - 1); w0 = min(max(w0, 0), IW - 1);

    auto L = [&](int dd, int hh, int ww) -> float {
        return (float)in[((((long)b * ID + dd) * IH + hh) * IW + ww) * C + c];
    };
    float c00 = L(d0, h0, w0) * (1.f - fw) + L(d0, h0, w1) * fw;
    float c01 = L(d0, h1, w0) * (1.f - fw) + L(d0, h1, w1) * fw;
    float c10 = L(d1, h0, w0) * (1.f - fw) + L(d1, h0, w1) * fw;
    float c11 = L(d1, h1, w0) * (1.f - fw) + L(d1, h1, w1) * fw;
    float c0 = c00 * (1.f - fh) + c01 * fh;
    float c1 = c10 * (1.f - fh) + c11 * fh;
    float v  = c0 * (1.f - fd) + c1 * fd;
    v = fmaxf(v, 0.f);
    if (skip) v += (float)skip[i];
    out[i] = (_Float16)v;
}

// ---------------- ResNet helpers (tiny, f32 NCHW) ----------------
__global__ void conv1x1_k(const float* __restrict__ x, const float* __restrict__ w,
                          float* __restrict__ out, int B, int Cin, int Cout, int HW, long total)
{
    long i = (long)blockIdx.x * blockDim.x + threadIdx.x;
    if (i >= total) return;
    int hw = (int)(i % HW); long t = i / HW;
    int co = (int)(t % Cout); int b = (int)(t / Cout);
    float s = 0.f;
    for (int ci = 0; ci < Cin; ++ci)
        s += w[co * Cin + ci] * x[((long)b * Cin + ci) * HW + hw];
    out[i] = s;
}

__global__ void conv3x3_k(const float* __restrict__ x, const float* __restrict__ wt,
                          float* __restrict__ out, int B, int Cin, int Cout, int H, int W,
                          long total)
{
    long i = (long)blockIdx.x * blockDim.x + threadIdx.x;
    if (i >= total) return;
    int ww_ = (int)(i % W); long t = i / W;
    int hh  = (int)(t % H); t /= H;
    int co  = (int)(t % Cout); int b = (int)(t / Cout);
    float s = 0.f;
    for (int ci = 0; ci < Cin; ++ci) {
        const float* xp = x + ((long)b * Cin + ci) * H * W;
        const float* wp = wt + ((long)co * Cin + ci) * 9;
        for (int k1 = 0; k1 < 3; ++k1) {
            int ih = hh + k1 - 1; if (ih < 0 || ih >= H) continue;
            for (int k2 = 0; k2 < 3; ++k2) {
                int iw = ww_ + k2 - 1; if (iw < 0 || iw >= W) continue;
                s += wp[k1 * 3 + k2] * xp[ih * W + iw];
            }
        }
    }
    out[i] = s;
}

// BatchNorm (batch statistics) + optional residual add + optional relu.
// One block per channel; full reduce over B*HW then apply.
// outF!=null -> f32 NCHW; else write f16 into packed views buffer
// (B,64,64,96) at channel offset coff (transpose on the fly).
__global__ __launch_bounds__(256)
void bn_k(const float* __restrict__ x, const float* __restrict__ g,
          const float* __restrict__ bta, const float* __restrict__ addt,
          float* __restrict__ outF, _Float16* __restrict__ outH,
          int B, int C, int HW, int relu, int coff)
{
    const int c = blockIdx.x, tid = threadIdx.x;
    const long N = (long)B * HW;
    __shared__ float s1[256], s2[256];
    float sum = 0.f, sq = 0.f;
    for (long i = tid; i < N; i += 256) {
        int b = (int)(i / HW); int hw = (int)(i % HW);
        float v = x[((long)b * C + c) * HW + hw];
        sum += v; sq += v * v;
    }
    s1[tid] = sum; s2[tid] = sq;
    __syncthreads();
    for (int st = 128; st > 0; st >>= 1) {
        if (tid < st) { s1[tid] += s1[tid + st]; s2[tid] += s2[tid + st]; }
        __syncthreads();
    }
    const float mean = s1[0] / (float)N;
    const float var  = s2[0] / (float)N - mean * mean;
    const float inv  = rsqrtf(var + 1e-5f);
    const float gg = g[c], bb = bta[c];
    for (long i = tid; i < N; i += 256) {
        int b = (int)(i / HW); int hw = (int)(i % HW);
        long xi = ((long)b * C + c) * HW + hw;
        float v = (x[xi] - mean) * inv * gg + bb;
        if (addt) v += addt[xi];
        if (relu) v = fmaxf(v, 0.f);
        if (outF) outF[xi] = v;
        else      outH[((long)b * HW + hw) * 96 + coff + c] = (_Float16)v;
    }
}

// Gather per point, then fused 1x1x1 conv (m4) + softmax over 2 channels.
__global__ void gather_k(const _Float16* __restrict__ vol, const float* __restrict__ cad,
                         const float* __restrict__ m4w, const float* __restrict__ m4b,
                         float* __restrict__ out, int B, int Npt, int R)
{
    int i = blockIdx.x * blockDim.x + threadIdx.x;
    if (i >= B * Npt) return;
    int b = i / Npt;
    const float* cp = cad + (long)i * 3;
    int id[3];
#pragma unroll
    for (int k = 0; k < 3; ++k) {
        int v = (int)floorf((cp[k] + 1.f) * 0.5f * (float)R);
        id[k] = min(max(v, 0), R - 1);
    }
    long base = ((((long)b * R + id[0]) * R + id[1]) * R + id[2]) * 2;
    float v0 = (float)vol[base], v1 = (float)vol[base + 1];
    float y0 = m4w[0] * v0 + m4w[1] * v1 + m4b[0];
    float y1 = m4w[2] * v0 + m4w[3] * v1 + m4b[1];
    float mx = fmaxf(y0, y1);
    float e0 = expf(y0 - mx), e1 = expf(y1 - mx);
    float s = e0 + e1;
    out[(long)i * 2]     = e0 / s;
    out[(long)i * 2 + 1] = e1 / s;
}

// ---------------------------------------------------------------------------
extern "C" void kernel_launch(void* const* d_in, const int* in_sizes, int n_in,
                              void* d_out, int out_size, void* d_ws, size_t ws_size,
                              hipStream_t stream)
{
    (void)in_sizes; (void)n_in; (void)out_size; (void)ws_size;
    const int B = 2;
    const float* side  = (const float*)d_in[0];
    const float* front = (const float*)d_in[1];
    const float* topv  = (const float*)d_in[2];
    const float* cad   = (const float*)d_in[3];
    // params flattened in JAX tree order (dict keys sorted at each level)
    const float* r_b0 = (const float*)d_in[4];
    const float* r_b1 = (const float*)d_in[5];
    const float* r_b2 = (const float*)d_in[6];
    const float* r_bs = (const float*)d_in[7];
    const float* r_g0 = (const float*)d_in[8];
    const float* r_g1 = (const float*)d_in[9];
    const float* r_g2 = (const float*)d_in[10];
    const float* r_gs = (const float*)d_in[11];
    const float* r_w0 = (const float*)d_in[12];
    const float* r_w1 = (const float*)d_in[13];
    const float* r_w2 = (const float*)d_in[14];
    const float* r_ws = (const float*)d_in[15];
    const float* d2b = (const float*)d_in[16];
    const float* d2w = (const float*)d_in[17];
    const float* d3b = (const float*)d_in[18];
    const float* d3w = (const float*)d_in[19];
    const float* d4b = (const float*)d_in[20];
    const float* d4w = (const float*)d_in[21];
    const float* d5b = (const float*)d_in[22];
    const float* d5w = (const float*)d_in[23];
    const float* e1b = (const float*)d_in[24];
    const float* e1w = (const float*)d_in[25];
    const float* e2b = (const float*)d_in[26];
    const float* e2w = (const float*)d_in[27];
    const float* e3b = (const float*)d_in[28];
    const float* e3w = (const float*)d_in[29];
    const float* e4b = (const float*)d_in[30];
    const float* e4w = (const float*)d_in[31];
    const float* m4b = (const float*)d_in[32];
    const float* m4w = (const float*)d_in[33];
    float* outp = (float*)d_out;

    // -------- workspace bump allocator --------
    uintptr_t wsbase = (uintptr_t)d_ws;
    size_t off = 0;
    auto alloc = [&](size_t bytes) -> void* {
        off = (off + 255) & ~(size_t)255;
        void* p = (void*)(wsbase + off);
        off += bytes;
        return p;
    };

    void* region0 = alloc((size_t)36 << 20);  // resnet f32 bufs, later transient conv outs
    _Float16* v2d = (_Float16*)alloc((size_t)B * 4096 * 96 * 2);
    auto wpk_bytes = [](int Cout, int Cin) -> size_t {
        return (size_t)((Cout + 15) / 16) * 27 * (Cin / 32) * 512 * 2;
    };
    _Float16* pkE1 = (_Float16*)alloc(wpk_bytes(32, 96));
    _Float16* pkE2 = (_Float16*)alloc(wpk_bytes(64, 32));
    _Float16* pkE3 = (_Float16*)alloc(wpk_bytes(128, 64));
    _Float16* pkE4 = (_Float16*)alloc(wpk_bytes(256, 128));
    _Float16* pkD2 = (_Float16*)alloc(wpk_bytes(128, 256));
    _Float16* pkD3 = (_Float16*)alloc(wpk_bytes(64, 128));
    _Float16* pkD4 = (_Float16*)alloc(wpk_bytes(32, 64));
    _Float16* pkD5 = (_Float16*)alloc(wpk_bytes(2, 32));
    _Float16* t1   = (_Float16*)alloc((size_t)B * 32768 * 32 * 2);
    _Float16* t2   = (_Float16*)alloc((size_t)B * 4096 * 64 * 2);
    _Float16* t3   = (_Float16*)alloc((size_t)B * 512 * 128 * 2);
    _Float16* bott = (_Float16*)alloc((size_t)B * 64 * 256 * 2);
    _Float16* u3   = (_Float16*)alloc((size_t)B * 512 * 128 * 2);
    _Float16* u4   = (_Float16*)alloc((size_t)B * 4096 * 64 * 2);
    _Float16* u5   = (_Float16*)alloc((size_t)B * 32768 * 32 * 2);
    _Float16* vol  = (_Float16*)alloc((size_t)B * 262144 * 2 * 2);

    // resnet f32 scratch lives in region0 (reused as `trans` afterwards)
    float* bufA = (float*)region0;         // (B,64,64,64)
    float* bufX = bufA + 524288;
    float* bufB = bufX + 524288;           // (B,32,64,64)
    float* bufY = bufB + 262144;
    float* bufC = bufY + 262144;
    float* bufZ = bufC + 262144;
    float* bufS = bufZ + 262144;
    _Float16* trans = (_Float16*)region0;  // transient conv outputs (channel-last f16)

    // -------- ResNet blocks (one per view), write packed view features --------
    const float* views[3] = { side, front, topv };
    for (int v = 0; v < 3; ++v) {
        long t = (long)B * 64 * 4096;
        conv1x1_k<<<(int)((t + 255) / 256), 256, 0, stream>>>(views[v], r_w0, bufA, B, 3, 64, 4096, t);
        bn_k<<<64, 256, 0, stream>>>(bufA, r_g0, r_b0, nullptr, bufX, nullptr, B, 64, 4096, 1, 0);
        t = (long)B * 32 * 4096;
        conv3x3_k<<<(int)((t + 255) / 256), 256, 0, stream>>>(bufX, r_w1, bufB, B, 64, 32, 64, 64, t);
        bn_k<<<32, 256, 0, stream>>>(bufB, r_g1, r_b1, nullptr, bufY, nullptr, B, 32, 4096, 1, 0);
        conv3x3_k<<<(int)((t + 255) / 256), 256, 0, stream>>>(bufY, r_w2, bufC, B, 32, 32, 64, 64, t);
        bn_k<<<32, 256, 0, stream>>>(bufC, r_g2, r_b2, nullptr, bufZ, nullptr, B, 32, 4096, 0, 0);
        conv1x1_k<<<(int)((t + 255) / 256), 256, 0, stream>>>(bufX, r_ws, bufS, B, 64, 32, 4096, t);
        bn_k<<<32, 256, 0, stream>>>(bufS, r_gs, r_bs, bufZ, nullptr, v2d, B, 32, 4096, 1, v * 32);
    }

    // -------- pack all conv3d weights into WMMA A-fragment order --------
    auto packw = [&](const float* w, _Float16* pk, int Cout, int Cin) {
        long tot = (long)((Cout + 15) / 16) * 27 * (Cin / 32) * 512;
        pack_w_k<<<(int)((tot + 255) / 256), 256, 0, stream>>>(w, pk, Cout, Cin, tot);
    };
    packw(e1w, pkE1, 32, 96);   packw(e2w, pkE2, 64, 32);
    packw(e3w, pkE3, 128, 64);  packw(e4w, pkE4, 256, 128);
    packw(d2w, pkD2, 128, 256); packw(d3w, pkD3, 64, 128);
    packw(d4w, pkD4, 32, 64);   packw(d5w, pkD5, 2, 32);

    auto conv = [&](const _Float16* in, const _Float16* pk, const float* bias, _Float16* o,
                    int Cin, int Cout, int Dm, int relu, int vm) {
        int DHW = Dm * Dm * Dm;
        int MT  = (Cout + 15) / 16;
        if (vm) {                                   // e1: DHW=262144, NT=4
            dim3 g(DHW / 256, MT, B);
            conv3d_wmma_k<1, 4><<<g, 128, 0, stream>>>(in, pk, bias, o, Cin, Cout, Dm, Dm, Dm, relu);
        } else if ((DHW & 255) == 0) {
            dim3 g(DHW / 256, MT, B);
            conv3d_wmma_k<0, 4><<<g, 128, 0, stream>>>(in, pk, bias, o, Cin, Cout, Dm, Dm, Dm, relu);
        } else if ((DHW & 127) == 0) {
            dim3 g(DHW / 128, MT, B);
            conv3d_wmma_k<0, 2><<<g, 128, 0, stream>>>(in, pk, bias, o, Cin, Cout, Dm, Dm, Dm, relu);
        } else {                                    // d2: DHW=64
            dim3 g(DHW / 64, MT, B);
            conv3d_wmma_k<0, 1><<<g, 128, 0, stream>>>(in, pk, bias, o, Cin, Cout, Dm, Dm, Dm, relu);
        }
    };
    auto pool = [&](const _Float16* in, _Float16* o, int C, int OD) {
        long tot = (long)B * OD * OD * OD * C;
        maxpool_k<<<(int)((tot + 255) / 256), 256, 0, stream>>>(in, o, C, OD, OD, OD, tot);
    };
    auto up = [&](const _Float16* in, const _Float16* skip, _Float16* o, int C, int ID) {
        long tot = (long)B * (2 * ID) * (2 * ID) * (2 * ID) * C;
        up2_k<<<(int)((tot + 255) / 256), 256, 0, stream>>>(in, skip, o, C, ID, ID, ID, tot);
    };

    // -------- U-Net (e1 fuses the broadcast/concat via VIEWS mode) --------
    conv(v2d,  pkE1, e1b, trans, 96, 32, 64, 1, 1);  pool(trans, t1, 32, 32);
    conv(t1,   pkE2, e2b, trans, 32, 64, 32, 1, 0);  pool(trans, t2, 64, 16);
    conv(t2,   pkE3, e3b, trans, 64, 128, 16, 1, 0); pool(trans, t3, 128, 8);
    conv(t3,   pkE4, e4b, trans, 128, 256, 8, 1, 0); pool(trans, bott, 256, 4);
    conv(bott, pkD2, d2b, trans, 256, 128, 4, 0, 0); up(trans, t3, u3, 128, 4);
    conv(u3,   pkD3, d3b, trans, 128, 64, 8, 0, 0);  up(trans, t2, u4, 64, 8);
    conv(u4,   pkD4, d4b, trans, 64, 32, 16, 0, 0);  up(trans, t1, u5, 32, 16);
    conv(u5,   pkD5, d5b, trans, 32, 2, 32, 0, 0);   up(trans, nullptr, vol, 2, 32);

    // -------- gather + fused m4 conv + softmax --------
    gather_k<<<(B * 1024 + 255) / 256, 256, 0, stream>>>(vol, cad, m4w, m4b, outp, B, 1024, 64);
}
